// RegionalHead_20529943675114
// MI455X (gfx1250) — matrix operand order repaired and checked by reading, not compiled, not run on hardware
//
#include <hip/hip_runtime.h>
#include <hip/hip_bf16.h>

// Problem dims (fixed by the reference)
#define S_DIM 32
#define C_DIM 2048
#define L_DIM 256
#define H_DIM 128

typedef __attribute__((ext_vector_type(16))) _Float16 v16h;
typedef __attribute__((ext_vector_type(8)))  _Float16 v8h;
typedef __attribute__((ext_vector_type(8)))  float    v8f;
typedef __attribute__((ext_vector_type(8)))  float    f32x8;

static __device__ __forceinline__ v8f wmma_f16(v16h a, v16h b, v8f c) {
    return __builtin_amdgcn_wmma_f32_16x16x32_f16(false, a, false, b, (short)0, c, false, false);
}

// ---------------------------------------------------------------------------
// Kernel 0: one-shot weight conversion fp32 -> f16.  Wq is pre-scaled by
// 1/sqrt(H) so attention logits need no extra scaling anywhere.
// ---------------------------------------------------------------------------
__global__ __launch_bounds__(256) void wcvt_kernel(
    const float* __restrict__ Wk,
    const float* __restrict__ Wq,
    const float* __restrict__ Wv,
    _Float16* __restrict__ Whq,
    _Float16* __restrict__ Whk,
    _Float16* __restrict__ Whv)
{
    const float qscale = 0.08838834764831845f;   // 1/sqrt(128)
    const int i = blockIdx.x * 256 + threadIdx.x;
    if (i < H_DIM * L_DIM) {
        Whq[i] = (_Float16)(Wq[i] * qscale);
        Whk[i] = (_Float16)Wk[i];
        Whv[i] = (_Float16)Wv[i];
    }
}

// ---------------------------------------------------------------------------
// Kernel 1: QKV projection.  One wave computes one 16x128 output tile for one
// of {Q,K,V} for one s.  All 8 B-fragments of a K-step are staged into
// registers before the WMMA chain so loads clause together and WMMAs retire
// against partial loadcnt waits.  V is stored transposed (Vt[s][h][key]).
// ---------------------------------------------------------------------------
__global__ __launch_bounds__(256) void qkv_proj_kernel(
    const float*    __restrict__ x,
    const _Float16* __restrict__ Whq,
    const _Float16* __restrict__ Whk,
    const _Float16* __restrict__ Whv,
    _Float16* __restrict__ Qh,
    _Float16* __restrict__ Kh,
    _Float16* __restrict__ Vt)
{
    const int lane = threadIdx.x & 31;
    const int wave = threadIdx.x >> 5;
    const int gw   = blockIdx.x * 8 + wave;          // 12288 waves total
    const int s    = gw / (3 * (C_DIM / 16));
    const int rem  = gw % (3 * (C_DIM / 16));
    const int mat  = rem / (C_DIM / 16);             // 0=Q, 1=K, 2=V
    const int rb   = rem % (C_DIM / 16);
    const int m0   = rb * 16;

    const _Float16* Wh = (mat == 0) ? Whq : ((mat == 1) ? Whk : Whv);

    const int lrow = lane & 15;
    const int hl   = lane >> 4;        // lane-half select
    const int koff = hl * 16;          // A-frag K sub-offset

    v8f acc[8] = {};

    #pragma unroll
    for (int kb = 0; kb < L_DIM; kb += 32) {
        // Stage all 8 B fragments (f16 weights, 32B each) up front.
        v16h b[8];
        #pragma unroll
        for (int nt = 0; nt < 8; ++nt) {
            const int n  = nt * 16 + lrow;
            const int kk = kb + hl * 16;
            b[nt] = *(const v16h*)(Wh + (size_t)n * L_DIM + kk);
        }

        // A fragment: rows of x (fp32 -> f16)
        const float* xp = x + ((size_t)(s * C_DIM + m0 + lrow)) * L_DIM + kb + koff;
        f32x8 a0 = *(const f32x8*)xp;          // K = kb+koff   .. +7
        f32x8 a1 = *(const f32x8*)(xp + 8);    // K = kb+koff+8 .. +15
        v16h a;
        #pragma unroll
        for (int i = 0; i < 8; ++i) { a[i] = (_Float16)a0[i]; a[8 + i] = (_Float16)a1[i]; }

        #pragma unroll
        for (int nt = 0; nt < 8; ++nt)
            acc[nt] = wmma_f16(a, b[nt], acc[nt]);
    }

    #pragma unroll
    for (int nt = 0; nt < 8; ++nt) {
        #pragma unroll
        for (int r = 0; r < 8; ++r) {
            const int orow = m0 + r + 8 * hl;          // C/D layout: row = vgpr + 8*half
            const int ocol = nt * 16 + lrow;           // col = lane%16
            const _Float16 v = (_Float16)acc[nt][r];
            if (mat == 0) {
                Qh[((size_t)(s * C_DIM + orow)) * H_DIM + ocol] = v;
            } else if (mat == 1) {
                Kh[((size_t)(s * C_DIM + orow)) * H_DIM + ocol] = v;
            } else {
                Vt[(size_t)s * H_DIM * C_DIM + (size_t)ocol * C_DIM + orow] = v;
            }
        }
    }
}

// ---------------------------------------------------------------------------
// Kernel 2: flash attention.  One wave owns 16 query rows x full head (128),
// streams key tiles of 32, online softmax, P transposed via private LDS tile.
// K / V B-fragments are staged into register arrays before each WMMA chain.
// Per-s K+V working set is ~1MB f16 -> L2-resident across all 128 q-blocks.
// ---------------------------------------------------------------------------
__global__ __launch_bounds__(256) void attn_kernel(
    const _Float16* __restrict__ Qh,
    const _Float16* __restrict__ Kh,
    const _Float16* __restrict__ Vt,
    float* __restrict__ out)
{
    __shared__ _Float16 lds_p[8][16 * 32];   // 1 KB per wave, 8 KB total

    const int lane = threadIdx.x & 31;
    const int wave = threadIdx.x >> 5;
    const int s    = blockIdx.x >> 4;                    // 512 blocks = 32 s * 16
    const int qblk = (blockIdx.x & 15) * 8 + wave;       // 128 q-blocks per s
    const int q0   = qblk * 16;

    const int lrow = lane & 15;
    const int hl   = lane >> 4;
    const int koff = hl * 16;

    // Q fragments: A layout, 4 chunks of K=32 covering head dim 128
    v16h qa[4];
    #pragma unroll
    for (int c = 0; c < 4; ++c) {
        const _Float16* qp = Qh + ((size_t)(s * C_DIM + q0 + lrow)) * H_DIM + c * 32 + koff;
        v8h lo = *(const v8h*)qp;
        v8h hi = *(const v8h*)(qp + 8);
        #pragma unroll
        for (int i = 0; i < 8; ++i) { qa[c][i] = lo[i]; qa[c][8 + i] = hi[i]; }
    }

    v8f acc[8] = {};
    float m_r[8], l_r[8];
    #pragma unroll
    for (int r = 0; r < 8; ++r) { m_r[r] = -INFINITY; l_r[r] = 0.0f; }

    const _Float16* Ks = Kh + (size_t)s * C_DIM * H_DIM;
    const _Float16* Vs = Vt + (size_t)s * H_DIM * C_DIM;
    _Float16* pbuf = &lds_p[wave][0];

    for (int j = 0; j < C_DIM; j += 32) {
        // ---- stage the 8 K B-fragments for this 32-key tile ----
        v16h kf[8];
        #pragma unroll
        for (int c = 0; c < 4; ++c) {
            const int kk = c * 32 + hl * 16;
            kf[2 * c]     = *(const v16h*)(Ks + (size_t)(j + lrow) * H_DIM + kk);
            kf[2 * c + 1] = *(const v16h*)(Ks + (size_t)(j + 16 + lrow) * H_DIM + kk);
        }

        // ---- S = Q . K^T for 32 keys (two 16-col tiles) ----
        v8f s0 = {}, s1 = {};
        #pragma unroll
        for (int c = 0; c < 4; ++c) {
            s0 = wmma_f16(qa[c], kf[2 * c],     s0);
            s1 = wmma_f16(qa[c], kf[2 * c + 1], s1);
        }

        // ---- online softmax (row = vgpr + 8*half, cols across 16 lanes) ----
        float p0[8], p1[8], alpha[8];
        #pragma unroll
        for (int r = 0; r < 8; ++r) {
            float mx = fmaxf(s0[r], s1[r]);
            mx = fmaxf(mx, __shfl_xor(mx, 1, 32));
            mx = fmaxf(mx, __shfl_xor(mx, 2, 32));
            mx = fmaxf(mx, __shfl_xor(mx, 4, 32));
            mx = fmaxf(mx, __shfl_xor(mx, 8, 32));
            const float mnew = fmaxf(m_r[r], mx);
            const float al = __expf(m_r[r] - mnew);
            const float e0 = __expf(s0[r] - mnew);
            const float e1 = __expf(s1[r] - mnew);
            float rs = e0 + e1;
            rs += __shfl_xor(rs, 1, 32);
            rs += __shfl_xor(rs, 2, 32);
            rs += __shfl_xor(rs, 4, 32);
            rs += __shfl_xor(rs, 8, 32);
            l_r[r] = l_r[r] * al + rs;
            m_r[r] = mnew;
            p0[r] = e0; p1[r] = e1; alpha[r] = al;
        }
        #pragma unroll
        for (int nt = 0; nt < 8; ++nt)
            #pragma unroll
            for (int r = 0; r < 8; ++r)
                acc[nt][r] *= alpha[r];

        // ---- P: C-layout -> A-layout via LDS (16x32 f16 tile) ----
        #pragma unroll
        for (int r = 0; r < 8; ++r) {
            pbuf[(r + 8 * hl) * 32 + lrow]      = (_Float16)p0[r];
            pbuf[(r + 8 * hl) * 32 + 16 + lrow] = (_Float16)p1[r];
        }
        __syncthreads();
        v16h pa;
        {
            const _Float16* pp = pbuf + lrow * 32 + koff;
            v8h lo = *(const v8h*)pp;
            v8h hi = *(const v8h*)(pp + 8);
            #pragma unroll
            for (int i = 0; i < 8; ++i) { pa[i] = lo[i]; pa[8 + i] = hi[i]; }
        }

        // ---- out += P . V : stage 8 V B-fragments, then WMMA chain ----
        v16h vf[8];
        #pragma unroll
        for (int nt = 0; nt < 8; ++nt)
            vf[nt] = *(const v16h*)(Vs + (size_t)(nt * 16 + lrow) * C_DIM + j + hl * 16);
        #pragma unroll
        for (int nt = 0; nt < 8; ++nt)
            acc[nt] = wmma_f16(pa, vf[nt], acc[nt]);
        __syncthreads();
    }

    // ---- normalize and store fp32 ----
    float invl[8];
    #pragma unroll
    for (int r = 0; r < 8; ++r) invl[r] = 1.0f / l_r[r];
    #pragma unroll
    for (int nt = 0; nt < 8; ++nt) {
        #pragma unroll
        for (int r = 0; r < 8; ++r) {
            const int orow = q0 + r + 8 * hl;
            const int ocol = nt * 16 + lrow;
            out[((size_t)(s * C_DIM + orow)) * H_DIM + ocol] = acc[nt][r] * invl[r];
        }
    }
}

// ---------------------------------------------------------------------------
extern "C" void kernel_launch(void* const* d_in, const int* in_sizes, int n_in,
                              void* d_out, int out_size, void* d_ws, size_t ws_size,
                              hipStream_t stream) {
    (void)in_sizes; (void)n_in; (void)out_size; (void)ws_size;
    const float* x  = (const float*)d_in[0];
    const float* Wk = (const float*)d_in[1];
    const float* Wq = (const float*)d_in[2];
    const float* Wv = (const float*)d_in[3];
    float* out = (float*)d_out;

    // Workspace layout (all f16):
    //   Qh, Kh : [32][2048][128] row-major        (16 MB each)
    //   Vt     : [32][128][2048] transposed       (16 MB)
    //   Whq/Whk/Whv : [128][256]                  (64 KB each)
    _Float16* Qh  = (_Float16*)d_ws;
    _Float16* Kh  = Qh  + (size_t)S_DIM * C_DIM * H_DIM;
    _Float16* Vt  = Kh  + (size_t)S_DIM * C_DIM * H_DIM;
    _Float16* Whq = Vt  + (size_t)S_DIM * C_DIM * H_DIM;
    _Float16* Whk = Whq + (size_t)H_DIM * L_DIM;
    _Float16* Whv = Whk + (size_t)H_DIM * L_DIM;

    wcvt_kernel<<<128, 256, 0, stream>>>(Wk, Wq, Wv, Whq, Whk, Whv);
    qkv_proj_kernel<<<1536, 256, 0, stream>>>(x, Whq, Whk, Whv, Qh, Kh, Vt);
    attn_kernel<<<512, 256, 0, stream>>>(Qh, Kh, Vt, out);
}
